// KeywordWeighting_29188597744312
// MI455X (gfx1250) — compile-verified, matrix operand order checked by reading, not run on hardware
//
#include <hip/hip_runtime.h>

// Problem constants (from the reference)
constexpr int B  = 32;
constexpr int LV = 4096;
constexpr int LT = 64;
constexpr int D  = 256;
constexpr int C  = 128;

constexpr int SPLIT = 8;            // split-K blocks per batch for the scatter
constexpr int ROWS  = LV / SPLIT;   // 512 rows per block
constexpr int RS    = D + 2;        // padded LDS row stride for rep (bank-conflict avoidance)

typedef float    v2f  __attribute__((ext_vector_type(2)));
typedef float    v8f  __attribute__((ext_vector_type(8)));
typedef _Float16 v16h __attribute__((ext_vector_type(16)));

// ---------------------------------------------------------------------------
// Kernel 0: zero the workspace accumulators (ws is poisoned by the harness)
// ---------------------------------------------------------------------------
__global__ void kw_zero_ws(float* __restrict__ p, int n) {
    int i = blockIdx.x * blockDim.x + threadIdx.x;
    if (i < n) p[i] = 0.0f;
}

// ---------------------------------------------------------------------------
// Kernel 1: split-K scatter-sum of src_vid into per-cluster sums + counts.
// Whole per-batch accumulator (C*D = 128 KB) lives in LDS; thread tid owns
// dim d == tid exclusively, so the inner loop needs ZERO atomics.
// Partials are flushed with global f32 atomics.
// ---------------------------------------------------------------------------
__global__ void kw_scatter(const float* __restrict__ vid,
                           const int*   __restrict__ clust,
                           float* __restrict__ gsum,
                           float* __restrict__ gcount) {
    extern __shared__ float smem[];
    float* lsum   = smem;          // C*D floats
    float* lcount = smem + C * D;  // C floats

    const int tid = threadIdx.x;                 // 0..255  (== dim index d)
    const int b   = blockIdx.x / SPLIT;
    const int s   = blockIdx.x % SPLIT;

    for (int i = tid; i < C * D + C; i += 256) smem[i] = 0.0f;
    __syncthreads();

    const int l0 = s * ROWS;
    const float* vrow = vid + ((size_t)b * LV + l0) * D + tid;
    const int*   crow = clust + (size_t)b * LV + l0;

    for (int r = 0; r < ROWS; ++r) {
        int   c = crow[r];                       // block-uniform -> scalar load
        float v = vrow[(size_t)r * D];           // coalesced 128B per wave
        lsum[c * D + tid] += v;                  // private per-thread slot: no atomic
        if (tid == 0) lcount[c] += 1.0f;
    }
    __syncthreads();

    float* gs = gsum + (size_t)b * C * D;
    for (int i = tid; i < C * D; i += 256) atomicAdd(&gs[i], lsum[i]);
    if (tid < C) atomicAdd(&gcount[b * C + tid], lcount[tid]);
}

// ---------------------------------------------------------------------------
// Kernel 2 (fused): rep = sum/count into LDS, dot = txt @ rep^T via WMMA f32,
// masked softmax-max (== 1/sum(exp)), min-max normalize, write (B,LT).
// One block per batch; 8 waves = 8 N-tiles of 16 clusters.
// ---------------------------------------------------------------------------
__global__ void kw_rep_dot(const float* __restrict__ gsum,
                           const float* __restrict__ gcount,
                           const float* __restrict__ txt,
                           const float* __restrict__ tmask,
                           float* __restrict__ out) {
    extern __shared__ float smem[];
    float* rep = smem;                 // C * RS
    float* dot = rep + C * RS;         // LT * C
    float* cnt = dot + LT * C;         // C
    float* kw  = cnt + C;              // LT
    float* mm  = kw + LT;              // 2

    const int tid = threadIdx.x;
    const int b   = blockIdx.x;

    if (tid < C) cnt[tid] = gcount[b * C + tid];
    __syncthreads();

    const float* gs = gsum + (size_t)b * C * D;
    for (int i = tid; i < C * D; i += 256) {
        int   c  = i / D, d = i % D;
        float ct = cnt[c];
        rep[c * RS + d] = (ct > 0.0f) ? gs[i] / (ct + 1e-8f) : 0.0f;
    }
    __syncthreads();

    // ---- WMMA: dot[l, c] = sum_d txt[l,d] * rep[c,d] --------------------
    const int wave = tid >> 5;
    const int lane = tid & 31;
    const int half = lane >> 4;        // 0: K=0,1   1: K=2,3   (16x4 f32 A/B layout)
    const int l16  = lane & 15;
    const int nT   = wave;             // N tile (clusters)
    const float* trow = txt + (size_t)b * LT * D;

    for (int mT = 0; mT < LT / 16; ++mT) {
        v8f acc = {};
#if defined(__has_builtin) && __has_builtin(__builtin_amdgcn_wmma_f32_16x16x4_f32)
        const float* arow = trow + (mT * 16 + l16) * D + 2 * half;   // A: 16x4 (M,K)
        const float* brow = rep + (nT * 16 + l16) * RS + 2 * half;   // B: rep[n][k]
        #pragma unroll 4
        for (int k0 = 0; k0 < D; k0 += 4) {
            v2f a  = *(const v2f*)(arow + k0);
            v2f bb = *(const v2f*)(brow + k0);
            acc = __builtin_amdgcn_wmma_f32_16x16x4_f32(
                false, a, false, bb, (short)0, acc, false, false);
        }
#elif defined(__has_builtin) && __has_builtin(__builtin_amdgcn_wmma_f32_16x16x32_f16)
        // Fallback: f16 inputs, f32 accumulate (16-bit A 16x32 VGPR layout)
        const float* arow = trow + (mT * 16 + l16) * D;
        const float* brow = rep + (nT * 16 + l16) * RS;
        for (int k0 = 0; k0 < D; k0 += 32) {
            v16h a, bb;
            #pragma unroll
            for (int j = 0; j < 16; ++j) {
                int K = j + (j < 8 ? (half ? 8 : 0) : (half ? 16 : 8));
                a[j]  = (_Float16)arow[k0 + K];
                bb[j] = (_Float16)brow[k0 + K];
            }
            acc = __builtin_amdgcn_wmma_f32_16x16x32_f16(
                false, a, false, bb, (short)0, acc, false, false);
        }
#else
        // Host-pass placeholder (never executed on device)
        (void)half;
#endif
        // D layout: VGPR r -> M = r (lanes 0-15) / r+8 (lanes 16-31), N = lane16
        #pragma unroll
        for (int r = 0; r < 8; ++r)
            dot[(mT * 16 + r + 8 * half) * C + nT * 16 + l16] = acc[r];
    }
    __syncthreads();

    // ---- masked softmax-max + mask:  kw = mask / sum(exp(d - dmax)) -----
    if (tid < LT) {
        float m = -__builtin_inff();
        for (int c = 0; c < C; ++c)
            if (cnt[c] > 0.0f) m = fmaxf(m, dot[tid * C + c]);
        float s = 0.0f;
        if (m > -__builtin_inff())
            for (int c = 0; c < C; ++c)
                if (cnt[c] > 0.0f) s += __expf(dot[tid * C + c] - m);
        kw[tid] = (m > -__builtin_inff()) ? tmask[b * LT + tid] / s : 0.0f;
    }
    __syncthreads();

    if (tid == 0) {
        float mn = kw[0], mx = kw[0];
        for (int i = 1; i < LT; ++i) { mn = fminf(mn, kw[i]); mx = fmaxf(mx, kw[i]); }
        mm[0] = mn; mm[1] = mx;
    }
    __syncthreads();

    if (tid < LT) out[b * LT + tid] = (kw[tid] - mm[0]) / (mm[1] - mm[0]);
}

// ---------------------------------------------------------------------------
extern "C" void kernel_launch(void* const* d_in, const int* in_sizes, int n_in,
                              void* d_out, int out_size, void* d_ws, size_t ws_size,
                              hipStream_t stream) {
    const float* vid   = (const float*)d_in[0];
    const float* txt   = (const float*)d_in[1];
    const float* tmask = (const float*)d_in[2];
    const int*   clust = (const int*)d_in[3];
    float* out = (float*)d_out;

    float* gsum   = (float*)d_ws;                  // B*C*D floats (4 MB)
    float* gcount = gsum + (size_t)B * C * D;      // B*C floats

    const int nz = B * C * D + B * C;
    kw_zero_ws<<<(nz + 255) / 256, 256, 0, stream>>>(gsum, nz);

    kw_scatter<<<B * SPLIT, 256, (C * D + C) * sizeof(float), stream>>>(
        vid, clust, gsum, gcount);

    const size_t sh2 = (size_t)(C * RS + LT * C + C + LT + 2) * sizeof(float);
    kw_rep_dot<<<B, 256, sh2, stream>>>(gsum, gcount, txt, tmask, out);
}
